// GATLayer_42245298323500
// MI455X (gfx1250) — matrix operand order repaired and checked by reading, not compiled, not run on hardware
//
#include <hip/hip_runtime.h>

#define B_  8
#define N_  1024
#define D_  512
#define NH_ 8
#define DK_ 64

typedef __attribute__((ext_vector_type(16))) _Float16 v16h;
typedef __attribute__((ext_vector_type(8)))  _Float16 v8h;
typedef __attribute__((ext_vector_type(8)))  float    v8f;

struct APair { v8h lo; v8h hi; };
union  AFrag { v16h v; APair p; };

__device__ __forceinline__ v8f wmma16(v16h a, v16h b, v8f c) {
  // v_wmma_f32_16x16x32_f16: D = A(16x32 f16) * B(32x16 f16) + C(16x16 f32)
  return __builtin_amdgcn_wmma_f32_16x16x32_f16(false, a, false, b, (short)0, c,
                                                false, false);
}

#if __has_builtin(__builtin_amdgcn_s_wait_asynccnt)
#define WAIT_ASYNC(n) __builtin_amdgcn_s_wait_asynccnt(n)
#else
#define WAIT_ASYNC(n) asm volatile("s_wait_asynccnt %0" ::"i"(n) : "memory")
#endif

// async global->LDS 16-byte copy per lane (ASYNCcnt-tracked).
// LDS operand VGPR = wave-relative LDS byte offset = low 32 bits of the
// generic address of a __shared__ object.
__device__ __forceinline__ void cp16_async(_Float16* lds, const _Float16* g) {
  unsigned l = (unsigned)(unsigned long long)(void*)lds;
  asm volatile("global_load_async_to_lds_b128 %0, %1, off"
               ::"v"(l), "v"(g)
               : "memory");
}

// ---------------- prep: f32 -> f16 conversions ----------------

__global__ void convx_kernel(const float* __restrict__ x, _Float16* __restrict__ xh) {
  int i = blockIdx.x * blockDim.x + threadIdx.x;          // exact: 8192*512 threads
  xh[i] = (_Float16)x[i];
}

__global__ void convw_kernel(const float* __restrict__ W, _Float16* __restrict__ Wt) {
  int i = blockIdx.x * blockDim.x + threadIdx.x;          // exact: 512*512 threads
  int r = i >> 9, c = i & 511;
  Wt[(size_t)c * D_ + r] = (_Float16)W[(size_t)r * D_ + c];  // transpose + narrow
}

// ---------------- fused QKV projection ----------------

__global__ void proj_kernel(const _Float16* __restrict__ xh,
                            const _Float16* __restrict__ Wqt,
                            const _Float16* __restrict__ Wkt,
                            const _Float16* __restrict__ Wvt,
                            _Float16* __restrict__ Qh,
                            _Float16* __restrict__ Kh,
                            _Float16* __restrict__ Vt) {
  const int lane = threadIdx.x & 31;
  const int wv   = threadIdx.x >> 5;
  const int hl   = lane & 15;
  const int hi   = lane >> 4;
  const int mt   = blockIdx.x * 4 + wv;      // 0..511
  const int row0 = mt * 16;
  const int c0   = blockIdx.y * 64;
  const int z    = blockIdx.z;
  const _Float16* W = (z == 0) ? Wqt : (z == 1) ? Wkt : Wvt;

  v8f acc[4] = {};
  const _Float16* arow = xh + (size_t)(row0 + hl) * D_;

  for (int k = 0; k < D_; k += 32) {
    AFrag a;
    a.v = *(const v16h*)(arow + k + hi * 16);
#pragma unroll
    for (int j = 0; j < 4; ++j) {
      v16h bb = *(const v16h*)(W + (size_t)(c0 + 16 * j + hl) * D_ + k + hi * 16);
      acc[j] = wmma16(a.v, bb, acc[j]);
    }
  }

  const int b  = row0 >> 10;
  const int n0 = row0 & 1023;
  const int h  = c0 >> 6;
  if (z < 2) {                                        // Q,K: [B,NH,N,DK]
    _Float16* dst = ((z == 0) ? Qh : Kh) + (((size_t)b * NH_ + h) * N_ + n0) * DK_;
#pragma unroll
    for (int j = 0; j < 4; ++j)
#pragma unroll
      for (int r = 0; r < 8; ++r)
        dst[(size_t)(r + 8 * hi) * DK_ + 16 * j + hl] = (_Float16)acc[j][r];
  } else {                                            // V transposed: [B,NH,DK,N]
    _Float16* dst = Vt + ((size_t)b * NH_ + h) * DK_ * N_;
#pragma unroll
    for (int j = 0; j < 4; ++j)
#pragma unroll
      for (int r = 0; r < 8; ++r)
        dst[(size_t)(16 * j + hl) * N_ + n0 + r + 8 * hi] = (_Float16)acc[j][r];
  }
}

// ---------------- fused masked flash attention ----------------
// wave: one (b, h, 16-query tile); key blocks of 32, K/V double-buffered in LDS
// via async global->LDS loads so WMMA overlaps the next tile's fill.

__global__ void attn_kernel(const _Float16* __restrict__ Qh,
                            const _Float16* __restrict__ Kh,
                            const _Float16* __restrict__ Vt,
                            const int* __restrict__ adj,
                            _Float16* __restrict__ Ah) {
  __shared__ __align__(32) _Float16 Kbuf[4][2][32 * 64];  // rows=key, cols=d
  __shared__ __align__(32) _Float16 Vbuf[4][2][64 * 32];  // rows=d,   cols=key
  __shared__ __align__(32) _Float16 pl[4][16 * 32];       // P tile

  const int lane = threadIdx.x & 31;
  const int wv   = threadIdx.x >> 5;
  const int hl   = lane & 15;
  const int hi   = lane >> 4;
  const int id   = blockIdx.x * 4 + wv;               // 0..4095
  const int qt   = id & 63;
  const int h    = (id >> 6) & 7;
  const int b    = id >> 9;

  const size_t hb = (size_t)b * NH_ + h;
  const _Float16* qrow = Qh + (hb * N_ + qt * 16 + hl) * DK_;
  AFrag qa0, qa1;
  qa0.v = *(const v16h*)(qrow + hi * 16);
  qa1.v = *(const v16h*)(qrow + 32 + hi * 16);

  const _Float16* kgbase = Kh + hb * N_ * DK_;                   // + (kb+lane)*DK
  const _Float16* vgbase = Vt + (hb * DK_ + lane) * (size_t)N_;  // d rows lane, lane+32
  const int* adjb = adj + ((size_t)b * N_ + qt * 16 + 8 * hi) * N_ + hl;

  float m_run[8], l_run[8];
  v8f acc[4] = {};
#pragma unroll
  for (int r = 0; r < 8; ++r) { m_run[r] = -1e30f; l_run[r] = 0.f; }

  const float scale = 0.125f;                         // 1/sqrt(DK)

  // stage one 32-key block (K: 32x64 f16, V: 64x32 f16) = 16 async b128 per lane-set
  auto stage = [&](int bufi, int kb) {
    const _Float16* kg = kgbase + (size_t)(kb + lane) * DK_;
    _Float16* kl = &Kbuf[wv][bufi][lane * 64];
#pragma unroll
    for (int i = 0; i < 8; ++i) cp16_async(kl + i * 8, kg + i * 8);
    const _Float16* vg = vgbase + kb;
    _Float16* vl = &Vbuf[wv][bufi][lane * 32];
#pragma unroll
    for (int i = 0; i < 4; ++i) cp16_async(vl + i * 8, vg + i * 8);
#pragma unroll
    for (int i = 0; i < 4; ++i)
      cp16_async(vl + 32 * 32 + i * 8, vg + (size_t)32 * N_ + i * 8);
  };

  int buf = 0;
  stage(0, 0);

  for (int kb = 0; kb < N_; kb += 32) {
    if (kb + 32 < N_) {                  // uniform branch: EXEC untouched
      stage(buf ^ 1, kb + 32);           // 16 more async ops in flight
      WAIT_ASYNC(16);                    // in-order: current buffer has landed
    } else {
      WAIT_ASYNC(0);
    }

    const _Float16* kt = &Kbuf[wv][buf][0];
    const _Float16* vt = &Vbuf[wv][buf][0];

    // ---- scores: two 16x16 tiles over 32 keys (DK=64 -> 2 k-steps each)
    v16h b00 = *(const v16h*)(kt + hl * 64 + hi * 16);
    v16h b01 = *(const v16h*)(kt + hl * 64 + 32 + hi * 16);
    v16h b10 = *(const v16h*)(kt + (16 + hl) * 64 + hi * 16);
    v16h b11 = *(const v16h*)(kt + (16 + hl) * 64 + 32 + hi * 16);
    v8f s0 = {}, s1 = {};
    s0 = wmma16(qa0.v, b00, s0);
    s0 = wmma16(qa1.v, b01, s0);
    s1 = wmma16(qa0.v, b10, s1);
    s1 = wmma16(qa1.v, b11, s1);

    // ---- mask + online softmax (C/D layout: VGPR r -> rows r / r+8)
#pragma unroll
    for (int r = 0; r < 8; ++r) {
      int a0 = adjb[(size_t)r * N_ + kb];
      int a1 = adjb[(size_t)r * N_ + kb + 16];
      float x0 = a0 ? s0[r] * scale : -3e30f;
      float x1 = a1 ? s1[r] * scale : -3e30f;
      float rm = fmaxf(x0, x1);
      rm = fmaxf(rm, __shfl_xor(rm, 1, 32));
      rm = fmaxf(rm, __shfl_xor(rm, 2, 32));
      rm = fmaxf(rm, __shfl_xor(rm, 4, 32));
      rm = fmaxf(rm, __shfl_xor(rm, 8, 32));
      float mn = fmaxf(m_run[r], rm);
      float f  = __expf(m_run[r] - mn);
      m_run[r] = mn;
      float p0 = __expf(x0 - mn);
      float p1 = __expf(x1 - mn);
      float ps = p0 + p1;
      ps += __shfl_xor(ps, 1, 32);
      ps += __shfl_xor(ps, 2, 32);
      ps += __shfl_xor(ps, 4, 32);
      ps += __shfl_xor(ps, 8, 32);
      l_run[r] = l_run[r] * f + ps;
#pragma unroll
      for (int j = 0; j < 4; ++j) acc[j][r] *= f;
      const int rr = r + 8 * hi;
      pl[wv][rr * 32 + hl]      = (_Float16)p0;
      pl[wv][rr * 32 + 16 + hl] = (_Float16)p1;
    }

    // wave-local LDS RAW: DS pipe is in-order; wait + compiler barrier
    asm volatile("s_wait_dscnt 0" ::: "memory");

    // ---- P(16x32) x V(32x64): P as A-fragment from LDS, V from staged LDS
    AFrag pa;
    const _Float16* pw = &pl[wv][0];
    pa.v = *(const v16h*)(pw + hl * 32 + hi * 16);
#pragma unroll
    for (int j = 0; j < 4; ++j) {
      v16h bv = *(const v16h*)(vt + (16 * j + hl) * 32 + hi * 16);
      acc[j] = wmma16(pa.v, bv, acc[j]);
    }
    buf ^= 1;
  }

  // ---- normalize, ReLU, store f16 concat-head layout [B,N,D]
  _Float16* orow = Ah + ((size_t)b * N_ + qt * 16 + 8 * hi) * D_ + h * DK_ + hl;
#pragma unroll
  for (int j = 0; j < 4; ++j)
#pragma unroll
    for (int r = 0; r < 8; ++r) {
      float ov = acc[j][r] / l_run[r];
      ov = fmaxf(ov, 0.f);
      orow[(size_t)r * D_ + 16 * j] = (_Float16)ov;
    }
}

// ---------------- output projection: relu(out) @ Wo -> f32 ----------------

__global__ void out_kernel(const _Float16* __restrict__ Ah,
                           const _Float16* __restrict__ Wot,
                           float* __restrict__ out) {
  const int lane = threadIdx.x & 31;
  const int wv   = threadIdx.x >> 5;
  const int hl   = lane & 15;
  const int hi   = lane >> 4;
  const int mt   = blockIdx.x * 4 + wv;
  const int row0 = mt * 16;
  const int c0   = blockIdx.y * 64;

  v8f acc[4] = {};
  const _Float16* arow = Ah + (size_t)(row0 + hl) * D_;
  for (int k = 0; k < D_; k += 32) {
    AFrag a;
    a.v = *(const v16h*)(arow + k + hi * 16);
#pragma unroll
    for (int j = 0; j < 4; ++j) {
      v16h bb = *(const v16h*)(Wot + (size_t)(c0 + 16 * j + hl) * D_ + k + hi * 16);
      acc[j] = wmma16(a.v, bb, acc[j]);
    }
  }
#pragma unroll
  for (int j = 0; j < 4; ++j)
#pragma unroll
    for (int r = 0; r < 8; ++r)
      out[(size_t)(row0 + r + 8 * hi) * D_ + c0 + 16 * j + hl] = acc[j][r];
}

// ---------------- launch ----------------

extern "C" void kernel_launch(void* const* d_in, const int* in_sizes, int n_in,
                              void* d_out, int out_size, void* d_ws, size_t ws_size,
                              hipStream_t stream) {
  const float* x   = (const float*)d_in[0];
  const int*   adj = (const int*)d_in[1];
  const float* Wq  = (const float*)d_in[2];
  const float* Wk  = (const float*)d_in[3];
  const float* Wv  = (const float*)d_in[4];
  const float* Wo  = (const float*)d_in[5];
  float* out = (float*)d_out;

  char* ws = (char*)d_ws;
  _Float16* xh  = (_Float16*)(ws + 0);          //  8,388,608  x as f16
  _Float16* Wqt = (_Float16*)(ws + 8388608);    //    524,288  Wq^T f16
  _Float16* Wkt = (_Float16*)(ws + 8912896);
  _Float16* Wvt = (_Float16*)(ws + 9437184);
  _Float16* Wot = (_Float16*)(ws + 9961472);
  _Float16* Qh  = (_Float16*)(ws + 10485760);   //  8,388,608  [B,NH,N,DK]
  _Float16* Kh  = (_Float16*)(ws + 18874368);   //  8,388,608  [B,NH,N,DK]
  _Float16* Vt  = (_Float16*)(ws + 27262976);   //  8,388,608  [B,NH,DK,N]
  _Float16* Ah  = (_Float16*)(ws + 35651584);   //  8,388,608  [B,N,D] relu'd

  convx_kernel<<<16384, 256, 0, stream>>>(x, xh);
  convw_kernel<<<1024, 256, 0, stream>>>(Wq, Wqt);
  convw_kernel<<<1024, 256, 0, stream>>>(Wk, Wkt);
  convw_kernel<<<1024, 256, 0, stream>>>(Wv, Wvt);
  convw_kernel<<<1024, 256, 0, stream>>>(Wo, Wot);

  proj_kernel<<<dim3(128, 8, 3), 128, 0, stream>>>(xh, Wqt, Wkt, Wvt, Qh, Kh, Vt);
  attn_kernel<<<1024, 128, 0, stream>>>(Qh, Kh, Vt, adj, Ah);
  out_kernel<<<dim3(128, 8), 128, 0, stream>>>(Ah, Wot, out);
}
